// Net_80942953660480
// MI455X (gfx1250) — matrix operand order repaired and checked by reading, not compiled, not run on hardware
//
#include <hip/hip_runtime.h>
#include <hip/hip_bf16.h>

// ---------------------------------------------------------------------------
// GINEConv x2 + BN + MLP head for MI455X (gfx1250).
//  - Edge scatter: f32 atomics; aggr buffers (51MB/19MB) are L2-resident (192MB).
//  - Node MLPs: v_wmma_f32_16x16x32_f16, f16 operands / f32 accumulate.
//    Weights are pre-swizzled into B-fragment layout so every WMMA operand is
//    loaded with ds_load_b128 / global_load_b128 (no scalar gathers).
//  - All K/N tile extents are template constants -> fully unrolled, no
//    divergent bounds predication around loads.
// ---------------------------------------------------------------------------

typedef _Float16 v16h __attribute__((ext_vector_type(16)));
typedef _Float16 v8h  __attribute__((ext_vector_type(8)));
typedef float    v8f  __attribute__((ext_vector_type(8)));

#define BN_EPS 1e-5f

// ---- WMMA fragment packing (CDNA5 ISA 7.12.2 layouts, wave32) -------------

// A (16x32 f16) from f32 global rows (optionally summed pair):
// lane L<16: row M=L, K = {kc+0..7, kc+16..23}; lane L>=16: row M=L-16,
// K = {kc+8..15, kc+24..31}. kc is compile-time (unrolled), KIN multiple of 16,
// so each 8-float group is uniformly in- or out-of-bounds.
template <int KIN, bool HAS2>
__device__ __forceinline__ v16h pack_a_global(const float* __restrict__ r1,
                                              const float* __restrict__ r2,
                                              int kc, int k0) {
  v16h a;
#pragma unroll
  for (int g = 0; g < 2; ++g) {
    float4 lo = make_float4(0.f, 0.f, 0.f, 0.f);
    float4 hi = make_float4(0.f, 0.f, 0.f, 0.f);
    if (kc + g * 16 + 16 <= KIN) {  // compile-time after unroll
      const float* p = r1 + kc + g * 16 + k0;
      lo = *(const float4*)(p);
      hi = *(const float4*)(p + 4);
      if (HAS2) {
        const float* q = r2 + kc + g * 16 + k0;
        float4 l2 = *(const float4*)(q);
        float4 h2 = *(const float4*)(q + 4);
        lo.x += l2.x; lo.y += l2.y; lo.z += l2.z; lo.w += l2.w;
        hi.x += h2.x; hi.y += h2.y; hi.z += h2.z; hi.w += h2.w;
      }
    }
    a[g * 8 + 0] = (_Float16)lo.x; a[g * 8 + 1] = (_Float16)lo.y;
    a[g * 8 + 2] = (_Float16)lo.z; a[g * 8 + 3] = (_Float16)lo.w;
    a[g * 8 + 4] = (_Float16)hi.x; a[g * 8 + 5] = (_Float16)hi.y;
    a[g * 8 + 6] = (_Float16)hi.z; a[g * 8 + 7] = (_Float16)hi.w;
  }
  return a;
}

// A from the f16 LDS staging tile (row stride 64 halves, zero-padded K):
// two contiguous 8-half runs -> 2x ds_load_b128.
__device__ __forceinline__ v16h pack_a_mid(const _Float16* row, int kc, int k0) {
  union { v16h v; v8h h[2]; } u;
  u.h[0] = *(const v8h*)(row + kc + k0);
  u.h[1] = *(const v8h*)(row + kc + 16 + k0);
  return u.v;
}

// B (32x16 f16) from swizzled LDS weights. Swizzled layout:
//   w_sw[(K/32)*48*32 + col*32 + (K%32)] = W[K][col]
// so lane (col, koff) reads two 8-half runs at koff and koff+16 -> 2x ds_load_b128.
__device__ __forceinline__ v16h pack_b_sw(const _Float16* wsw,
                                          int kc, int koff, int col) {
  const _Float16* p = wsw + ((kc >> 5) * (48 * 32)) + col * 32 + koff;
  union { v16h v; v8h h[2]; } u;
  u.h[0] = *(const v8h*)(p);
  u.h[1] = *(const v8h*)(p + 16);
  return u.v;
}

// ---- fused two-layer MLP on one 16-row tile (one wave) --------------------
// out = epilogue( relu(P @ W1 + b1) @ W2 + b2 ),  P = in1 (+ in2)
template <int KIN, int KP1, int KP2, int NT2, int OUT_COLS,
          bool HAS2, bool RELU_OUT, bool BN>
__device__ __forceinline__ void mlp_tile(
    const float* __restrict__ in1, const float* __restrict__ in2, int in_ld,
    const _Float16* sW1, const float* __restrict__ b1,
    const _Float16* sW2, const float* __restrict__ b2,
    const float* __restrict__ bnscale, const float* __restrict__ bnshift,
    float* __restrict__ out, int out_ld, _Float16* mid, int row0)
{
  const int lane = threadIdx.x & 31;
  const int nsel = lane & 15;
  const int hi   = (lane >> 4) & 1;
  const int k0   = hi * 8;
  const int koff = hi * 8;

  // zero 16x64 staging tile so padded K reads 0 (4x ds_store_b128 per lane)
  {
    v8h z = {};
#pragma unroll
    for (int i = 0; i < 4; ++i) ((v8h*)mid)[lane + 32 * i] = z;
  }

  const float* r1 = in1 + (long long)(row0 + nsel) * in_ld;
  const float* r2 = HAS2 ? in2 + (long long)(row0 + nsel) * in_ld : nullptr;

  // GEMM1: [16 x KIN] @ [KIN x 48] -> 3 N-tiles, f32 accumulate
  v8f acc[3] = {};
#pragma unroll
  for (int kc = 0; kc < KP1; kc += 32) {
    v16h a = pack_a_global<KIN, HAS2>(r1, r2, kc, k0);
#pragma unroll
    for (int t = 0; t < 3; ++t) {
      v16h b = pack_b_sw(sW1, kc, koff, t * 16 + nsel);
      acc[t] = __builtin_amdgcn_wmma_f32_16x16x32_f16(
          false, a, false, b, (short)0, acc[t], false, false);
    }
  }

  // bias + ReLU -> f16 staging (D layout: VGPR r -> row hi*8+r, col=t*16+nsel)
#pragma unroll
  for (int r = 0; r < 8; ++r) {
    int mrow = hi * 8 + r;
#pragma unroll
    for (int t = 0; t < 3; ++t) {
      int col = t * 16 + nsel;
      float v = acc[t][r] + b1[col];
      v = fmaxf(v, 0.0f);
      mid[mrow * 64 + col] = (_Float16)v;
    }
  }
  // same-wave DS store->load is processed in order by the LDS unit (ISA 7.3)

  // GEMM2: [16 x KP2] @ [KP2 x OUT] -> NT2 N-tiles
  const _Float16* arow = mid + nsel * 64;
  v8f acc2[NT2];
#pragma unroll
  for (int t = 0; t < NT2; ++t) acc2[t] = (v8f){};
#pragma unroll
  for (int kc = 0; kc < KP2; kc += 32) {
    v16h a = pack_a_mid(arow, kc, k0);
#pragma unroll
    for (int t = 0; t < NT2; ++t) {
      v16h b = pack_b_sw(sW2, kc, koff, t * 16 + nsel);
      acc2[t] = __builtin_amdgcn_wmma_f32_16x16x32_f16(
          false, a, false, b, (short)0, acc2[t], false, false);
    }
  }

  // epilogue: bias, optional ReLU, optional folded BN, masked store
#pragma unroll
  for (int r = 0; r < 8; ++r) {
    int row = row0 + hi * 8 + r;
#pragma unroll
    for (int t = 0; t < NT2; ++t) {
      int col = t * 16 + nsel;
      float v = acc2[t][r] + b2[col];
      if (RELU_OUT) v = fmaxf(v, 0.0f);
      if (BN)       v = v * bnscale[col] + bnshift[col];
      if (col < OUT_COLS) out[(long long)row * out_ld + col] = v;
    }
  }
}

// ---- prep: zero aggr, swizzle+convert weights, fold BN --------------------

// Write W[K][col] (f32, possibly padded) into f16 fragment layout:
//   d[(K/32)*48*32 + col*32 + (K%32)]
__device__ __forceinline__ void swz_cvt(long long gid, long long stride,
                                        const float* s, int srows, int scols,
                                        int sld, int kp, _Float16* d) {
  const long long n = (long long)(kp >> 5) * 48 * 32;
  for (long long i = gid; i < n; i += stride) {
    int chunk = (int)(i / (48 * 32));
    int rem   = (int)(i % (48 * 32));
    int col   = rem / 32;
    int kk    = rem % 32;
    int K     = chunk * 32 + kk;
    float v = (K < srows && col < scols) ? s[(long long)K * sld + col] : 0.0f;
    d[i] = (_Float16)v;
  }
}

__global__ void prep_kernel(
    float* zero_base, long long zero_n,
    const float* c1W1, const float* c1W2,
    const float* c2W1, const float* c2W2,
    const float* fc1W, const float* fc2W,
    _Float16* w1a, _Float16* w1b, _Float16* w2a, _Float16* w2b,
    _Float16* wf1, _Float16* wf2,
    const float* g1, const float* be1, const float* m1, const float* v1,
    const float* g2, const float* be2, const float* m2, const float* v2,
    float* bns1, float* bnh1, float* bns2, float* bnh2)
{
  long long gid    = (long long)blockIdx.x * blockDim.x + threadIdx.x;
  long long stride = (long long)gridDim.x * blockDim.x;
  for (long long i = gid; i < zero_n; i += stride) zero_base[i] = 0.0f;
  swz_cvt(gid, stride, c1W1, 128, 48, 48, 128, w1a);
  swz_cvt(gid, stride, c1W2,  48, 48, 48,  64, w1b);
  swz_cvt(gid, stride, c2W1,  48, 48, 48,  64, w2a);
  swz_cvt(gid, stride, c2W2,  48, 48, 48,  64, w2b);
  swz_cvt(gid, stride, fc1W,  48, 48, 48,  64, wf1);
  swz_cvt(gid, stride, fc2W,  48, 18, 18,  64, wf2);
  for (long long i = gid; i < 48; i += stride) {
    float s1 = g1[i] * rsqrtf(v1[i] + BN_EPS);
    bns1[i] = s1; bnh1[i] = be1[i] - m1[i] * s1;
    float s2 = g2[i] * rsqrtf(v2[i] + BN_EPS);
    bns2[i] = s2; bnh2[i] = be2[i] - m2[i] * s2;
  }
}

// ---- edge scatter: aggr[dst] += relu(feat[src]) ---------------------------
// One float4 per work-item; atomics land in the 192MB L2 (aggr is resident).
template <int LD4>
__global__ void scatter_relu4_kernel(const float* __restrict__ feat,
                                     const long long* __restrict__ src,
                                     const long long* __restrict__ dst,
                                     float* __restrict__ aggr,
                                     long long n_items)
{
  long long idx = (long long)blockIdx.x * blockDim.x + threadIdx.x;
  if (idx >= n_items) return;
  long long e = idx / LD4;
  int q = (int)(idx - e * LD4);
  long long s = src[e], d = dst[e];
  const float4 v = *(const float4*)(feat + s * (LD4 * 4) + q * 4);
  float* ap = aggr + d * (LD4 * 4) + q * 4;
  atomicAdd(ap + 0, fmaxf(v.x, 0.0f));
  atomicAdd(ap + 1, fmaxf(v.y, 0.0f));
  atomicAdd(ap + 2, fmaxf(v.z, 0.0f));
  atomicAdd(ap + 3, fmaxf(v.w, 0.0f));
}

// ---- MLP kernels ----------------------------------------------------------

__global__ void __launch_bounds__(256) mlp1_kernel(
    const float* __restrict__ x, const float* __restrict__ aggr,
    const _Float16* __restrict__ gW1, const _Float16* __restrict__ gW2,
    const float* __restrict__ b1, const float* __restrict__ b2,
    const float* __restrict__ bns, const float* __restrict__ bnh,
    float* __restrict__ h_out, int n_nodes)
{
  __shared__ __align__(16) _Float16 sW1[4 * 48 * 32];   // K=128
  __shared__ __align__(16) _Float16 sW2[2 * 48 * 32];   // K=64 (padded)
  __shared__ __align__(16) _Float16 sMid[8][16 * 64];
  const int tid = threadIdx.x;
  for (int i = tid; i < 6144 / 8; i += 256) ((uint4*)sW1)[i] = ((const uint4*)gW1)[i];
  for (int i = tid; i < 3072 / 8; i += 256) ((uint4*)sW2)[i] = ((const uint4*)gW2)[i];
  __syncthreads();
  const int wave = tid >> 5;
  const int row0 = (blockIdx.x * 8 + wave) * 16;
  if (row0 >= n_nodes) return;
  mlp_tile<128, 128, 64, 3, 48, true, true, true>(
      x, aggr, 128, sW1, b1, sW2, b2, bns, bnh, h_out, 48, sMid[wave], row0);
}

__global__ void __launch_bounds__(256) mlp2_kernel(
    const float* __restrict__ h1, const float* __restrict__ aggr,
    const _Float16* __restrict__ gW1, const _Float16* __restrict__ gW2,
    const float* __restrict__ b1, const float* __restrict__ b2,
    const float* __restrict__ bns, const float* __restrict__ bnh,
    float* __restrict__ h_out, int n_nodes)
{
  __shared__ __align__(16) _Float16 sW1[2 * 48 * 32];
  __shared__ __align__(16) _Float16 sW2[2 * 48 * 32];
  __shared__ __align__(16) _Float16 sMid[8][16 * 64];
  const int tid = threadIdx.x;
  for (int i = tid; i < 3072 / 8; i += 256) {
    ((uint4*)sW1)[i] = ((const uint4*)gW1)[i];
    ((uint4*)sW2)[i] = ((const uint4*)gW2)[i];
  }
  __syncthreads();
  const int wave = tid >> 5;
  const int row0 = (blockIdx.x * 8 + wave) * 16;
  if (row0 >= n_nodes) return;
  mlp_tile<48, 64, 64, 3, 48, true, true, true>(
      h1, aggr, 48, sW1, b1, sW2, b2, bns, bnh, h_out, 48, sMid[wave], row0);
}

__global__ void __launch_bounds__(256) head_kernel(
    const float* __restrict__ h2,
    const _Float16* __restrict__ gW1, const _Float16* __restrict__ gW2,
    const float* __restrict__ b1, const float* __restrict__ b2,
    float* __restrict__ out, int n_nodes)
{
  __shared__ __align__(16) _Float16 sW1[2 * 48 * 32];
  __shared__ __align__(16) _Float16 sW2[2 * 48 * 32];
  __shared__ __align__(16) _Float16 sMid[8][16 * 64];
  const int tid = threadIdx.x;
  for (int i = tid; i < 3072 / 8; i += 256) {
    ((uint4*)sW1)[i] = ((const uint4*)gW1)[i];
    ((uint4*)sW2)[i] = ((const uint4*)gW2)[i];
  }
  __syncthreads();
  const int wave = tid >> 5;
  const int row0 = (blockIdx.x * 8 + wave) * 16;
  if (row0 >= n_nodes) return;
  mlp_tile<48, 64, 64, 2, 18, false, false, false>(
      h2, nullptr, 48, sW1, b1, sW2, b2, nullptr, nullptr, out, 18,
      sMid[wave], row0);
}

// ---- host-side launcher ---------------------------------------------------

extern "C" void kernel_launch(void* const* d_in, const int* in_sizes, int n_in,
                              void* d_out, int out_size, void* d_ws, size_t ws_size,
                              hipStream_t stream) {
  (void)n_in; (void)out_size; (void)ws_size;
  const int       n_nodes = in_sizes[0] / 128;
  const long long n_edges = (long long)in_sizes[1] / 2;

  const float*     x     = (const float*)d_in[0];
  const long long* ei    = (const long long*)d_in[1];
  const long long* e_src = ei;
  const long long* e_dst = ei + n_edges;
  const float* c1W1 = (const float*)d_in[2];
  const float* c1b1 = (const float*)d_in[3];
  const float* c1W2 = (const float*)d_in[4];
  const float* c1b2 = (const float*)d_in[5];
  const float* c2W1 = (const float*)d_in[6];
  const float* c2b1 = (const float*)d_in[7];
  const float* c2W2 = (const float*)d_in[8];
  const float* c2b2 = (const float*)d_in[9];
  const float* bn1g = (const float*)d_in[10];
  const float* bn1b = (const float*)d_in[11];
  const float* bn1m = (const float*)d_in[12];
  const float* bn1v = (const float*)d_in[13];
  const float* bn2g = (const float*)d_in[14];
  const float* bn2b = (const float*)d_in[15];
  const float* bn2m = (const float*)d_in[16];
  const float* bn2v = (const float*)d_in[17];
  const float* fc1W = (const float*)d_in[18];
  const float* fc1b = (const float*)d_in[19];
  const float* fc2W = (const float*)d_in[20];
  const float* fc2b = (const float*)d_in[21];

  // workspace layout (aggr1|aggr2 contiguous for one-pass zeroing)
  float* aggr1 = (float*)d_ws;                               // N*128 f32
  float* aggr2 = aggr1 + (long long)n_nodes * 128;           // N*48  f32
  float* h1    = aggr2 + (long long)n_nodes * 48;            // N*48  f32
  float* h2    = h1    + (long long)n_nodes * 48;            // N*48  f32
  _Float16* w1a = (_Float16*)(h2 + (long long)n_nodes * 48); // 4 chunks swizzled
  _Float16* w1b = w1a + 4 * 48 * 32;                         // 2 chunks each
  _Float16* w2a = w1b + 2 * 48 * 32;
  _Float16* w2b = w2a + 2 * 48 * 32;
  _Float16* wf1 = w2b + 2 * 48 * 32;
  _Float16* wf2 = wf1 + 2 * 48 * 32;
  float* bns1 = (float*)(wf2 + 2 * 48 * 32);
  float* bnh1 = bns1 + 48;
  float* bns2 = bnh1 + 48;
  float* bnh2 = bns2 + 48;

  const long long zero_n = (long long)n_nodes * (128 + 48);
  prep_kernel<<<(int)((zero_n + 255) / 256), 256, 0, stream>>>(
      aggr1, zero_n, c1W1, c1W2, c2W1, c2W2, fc1W, fc2W,
      w1a, w1b, w2a, w2b, wf1, wf2,
      bn1g, bn1b, bn1m, bn1v, bn2g, bn2b, bn2m, bn2v,
      bns1, bnh1, bns2, bnh2);

  const long long it1 = n_edges * 32;  // 128 feats / 4 per item
  scatter_relu4_kernel<32><<<(int)((it1 + 255) / 256), 256, 0, stream>>>(
      x, e_src, e_dst, aggr1, it1);

  const int nblk = (n_nodes + 127) / 128;  // 8 waves x 16 rows per block
  mlp1_kernel<<<nblk, 256, 0, stream>>>(x, aggr1, w1a, w1b, c1b1, c1b2,
                                        bns1, bnh1, h1, n_nodes);

  const long long it2 = n_edges * 12;  // 48 feats / 4 per item
  scatter_relu4_kernel<12><<<(int)((it2 + 255) / 256), 256, 0, stream>>>(
      h1, e_src, e_dst, aggr2, it2);

  mlp2_kernel<<<nblk, 256, 0, stream>>>(h1, aggr2, w2a, w2b, c2b1, c2b2,
                                        bns2, bnh2, h2, n_nodes);

  head_kernel<<<nblk, 256, 0, stream>>>(h2, wf1, wf2, fc1b, fc2b,
                                        (float*)d_out, n_nodes);
}
